// MaskedAutoencoderViT_39221641347747
// MI455X (gfx1250) — matrix operand order repaired and checked by reading, not compiled, not run on hardware
//
#include <hip/hip_runtime.h>
#include <hip/hip_bf16.h>
#include <math.h>

// ---------------------------------------------------------------------------
// MI455X (gfx1250) RMT/MaSA ViT-MAE forward.
// - All GEMMs on v_wmma_f32_16x16x32_bf16 (bf16 in, f32 accumulate).
// - GEMM block tile 128x128, 8 waves x (32x64) -> 8 WMMA per 32-K step/wave.
// - A tiles staged via global_load_async_to_lds_b128 (ASYNCcnt); fragments
//   read as ds_load_b128 runs in contiguous k order (k = 16*half + s) for both A and B.
// - Attention kernels load WMMA fragments directly from global memory
//   (Q/K bf16 row-major, V bf16 transposed, probs bf16) - no LDS, no packing.
// ---------------------------------------------------------------------------

#define BB     8
#define CCH    2
#define IMGSZ  256
#define PSZ    16
#define DD     1024
#define LLAY   8
#define NHH    16
#define DFFF   4096
#define HPP    16
#define NTOK   256
#define SEQ    257
#define HDD    64
#define SPAD   288      // padded sequence (row strides stay 32B-aligned in bf16)
#define KPATCH 512
#define PRED   512

typedef __attribute__((ext_vector_type(16))) __bf16         v16bf;
typedef __attribute__((ext_vector_type(8)))  float          v8f;
typedef __attribute__((ext_vector_type(4)))  unsigned int   v4u;
typedef __attribute__((ext_vector_type(8)))  unsigned int   v8u;

static __device__ __forceinline__ unsigned short f2bf(float f) {
  unsigned int u = __float_as_uint(f);
  u += 0x7FFFu + ((u >> 16) & 1u);           // round-to-nearest-even
  return (unsigned short)(u >> 16);
}

// A-fragment: two contiguous 16B runs (k = 16*half.., k = 16*half+8..)
static __device__ __forceinline__ v16bf load_afrag(const unsigned short* row, int half) {
  v4u a0 = *(const v4u*)(row + half * 16);
  v4u a1 = *(const v4u*)(row + half * 16 + 8);
  return __builtin_bit_cast(v16bf, __builtin_shufflevector(a0, a1, 0, 1, 2, 3, 4, 5, 6, 7));
}
// B-fragment: one contiguous 32B run of 16 K-values
static __device__ __forceinline__ v16bf load_bfrag(const unsigned short* p) {
  return __builtin_bit_cast(v16bf, *(const v8u*)p);
}

static __device__ __forceinline__ v8f wmma_bf16(v16bf a, v16bf b, v8f c) {
  return __builtin_amdgcn_wmma_f32_16x16x32_bf16(false, a, false, b, (short)0, c, false, false);
}

// ---------------------------------------------------------------------------
// GEMM: out[M x Nn] = epi(Abf16[M x K] * Wf32[K x Nn] + bias)
// gelu: exact GELU; addres: += res;  outfmt 0: f32, 1: bf16, 2: bf16 vT-scatter
// Block tile 128x128, 8 waves x (32x64).
// ---------------------------------------------------------------------------
__global__ __launch_bounds__(256)
void gemm_bf16_wmma(const unsigned short* __restrict__ A, const float* __restrict__ W,
                    const float* __restrict__ bias, const float* __restrict__ res,
                    float* __restrict__ outf, unsigned short* __restrict__ outbf,
                    int M, int Nn, int K, int gelu, int addres, int outfmt)
{
  __shared__ __attribute__((aligned(16))) unsigned short As[128 * 40];   // rows of 32 bf16, 80B stride
  __shared__ __attribute__((aligned(16))) unsigned short BsF[8][32][24]; // fragment-order, 16 of 24 used

  const int tid  = threadIdx.x;
  const int lane = tid & 31;
  const int wv   = tid >> 5;
  const int wm   = wv & 3;                   // 4 row groups of 32
  const int wn   = wv >> 2;                  // 2 col groups of 64
  const int m0   = blockIdx.y * 128;
  const int n0   = blockIdx.x * 128;
  const int lrow = lane & 15;
  const int half = lane >> 4;
  const unsigned asbase = (unsigned)(size_t)As;

  v8f acc[2][4] = {};

  for (int kk = 0; kk < K; kk += 32) {
    // ---- async raw copy of A tile (128 x 32 bf16 = 512 x 16B chunks) ------
#pragma unroll
    for (int it = 0; it < 2; ++it) {
      const int chunk = tid + it * 256;
      const int r = chunk >> 2, seg = chunk & 3;
      const unsigned long long g =
          (unsigned long long)(size_t)(A + (size_t)(m0 + r) * K + kk + seg * 8);
      const unsigned dst = asbase + (unsigned)(r * 80 + seg * 16);
      asm volatile("global_load_async_to_lds_b128 %0, %1, off"
                   :: "v"(dst), "v"(g) : "memory");
    }
    // ---- stage W tile (32 x 128 f32) -> bf16 in B-fragment order ----------
#pragma unroll
    for (int it = 0; it < 4; ++it) {
      const int i = tid * 4 + it * 1024;               // 4096 elems, float4 each
      const int r = i >> 7, c = i & 127;               // r = k, c = n (coalesced)
      const float4 w4 = *(const float4*)&W[(size_t)(kk + r) * Nn + n0 + c];
      const int nt = c >> 4, lb = (c & 15) + 16 * (r >> 4), e = r & 15;
      BsF[nt][lb + 0][e] = f2bf(w4.x);
      BsF[nt][lb + 1][e] = f2bf(w4.y);
      BsF[nt][lb + 2][e] = f2bf(w4.z);
      BsF[nt][lb + 3][e] = f2bf(w4.w);
    }
    if (kk + 32 < K)
      __builtin_prefetch(&W[(size_t)(kk + 32 + (tid & 31)) * Nn + n0], 0, 1);
    asm volatile("s_wait_asynccnt 0x0" ::: "memory");
    __syncthreads();

    v16bf afr[2], bfr[4];
#pragma unroll
    for (int t = 0; t < 2; ++t)
      afr[t] = load_afrag(As + (wm * 32 + t * 16 + lrow) * 40, half);
#pragma unroll
    for (int j = 0; j < 4; ++j)
      bfr[j] = load_bfrag(&BsF[4 * wn + j][lane][0]);
#pragma unroll
    for (int i = 0; i < 2; ++i)
#pragma unroll
      for (int j = 0; j < 4; ++j)
        acc[i][j] = wmma_bf16(afr[i], bfr[j], acc[i][j]);
    __syncthreads();
  }

  // ---- epilogue: D layout row = v + 8*half, col = lane & 15 ---------------
#pragma unroll
  for (int i = 0; i < 2; ++i) {
#pragma unroll
    for (int j = 0; j < 4; ++j) {
      const int gc = n0 + wn * 64 + j * 16 + lrow;
      const float bc = bias[gc];
#pragma unroll
      for (int v = 0; v < 8; ++v) {
        const int gr = m0 + wm * 32 + i * 16 + v + 8 * half;
        float val = acc[i][j][v] + bc;
        if (addres) val += res[(size_t)gr * Nn + gc];
        if (gelu)   val = 0.5f * val * (1.0f + erff(val * 0.70710678f));
        if (outfmt == 0)      outf[(size_t)gr * Nn + gc] = val;
        else if (outfmt == 1) outbf[(size_t)gr * Nn + gc] = f2bf(val);
        else {                                   // V^T: [b, h, hd, token]
          const int bb = gr / SPAD, ss = gr % SPAD;
          const int hh = gc >> 6,  hd = gc & 63;
          outbf[(((size_t)bb * NHH + hh) * HDD + hd) * SPAD + ss] = f2bf(val);
        }
      }
    }
  }
}

// ---------------------------------------------------------------------------
// logits = 0.125 * Q.K^T + decay(h)*manhattan; fragments straight from global.
// One wave computes 16 queries x 32 keys.
// ---------------------------------------------------------------------------
__global__ __launch_bounds__(32)
void attn_logits_k(const unsigned short* __restrict__ q, const unsigned short* __restrict__ k,
                   float* __restrict__ logits)
{
  const int qt = blockIdx.x, kt2 = blockIdx.y, bh = blockIdx.z;
  const int b = bh / NHH, hh = bh % NHH;
  const int lane = threadIdx.x;
  const int lrow = lane & 15, half = lane >> 4;

  const unsigned short* qrow = q + ((size_t)b * SPAD + qt * 16 + lrow) * DD + hh * HDD;
  v8f acc[2] = {};
#pragma unroll
  for (int n = 0; n < 2; ++n) {                       // head dim 64 -> 2 steps
    const v16bf af = load_afrag(qrow + n * 32, half);
#pragma unroll
    for (int j = 0; j < 2; ++j) {
      const unsigned short* kp =
          k + ((size_t)b * SPAD + kt2 * 32 + j * 16 + lrow) * DD + hh * HDD + n * 32 + half * 16;
      acc[j] = wmma_bf16(af, load_bfrag(kp), acc[j]);
    }
  }
  const float decay = logf(1.0f - exp2f(-1.0f - (float)hh));
#pragma unroll
  for (int j = 0; j < 2; ++j)
#pragma unroll
    for (int v = 0; v < 8; ++v) {
      const int qi = qt * 16 + v + 8 * half;
      const int kj = kt2 * 32 + j * 16 + lrow;
      float m = 0.0f;
      if (qi > 0 && qi < SEQ && kj > 0 && kj < SEQ) {
        const int qy = (qi - 1) >> 4, qx = (qi - 1) & 15;
        const int ky = (kj - 1) >> 4, kx = (kj - 1) & 15;
        const int dy = qy > ky ? qy - ky : ky - qy;
        const int dx = qx > kx ? qx - kx : kx - qx;
        m = decay * (float)(dy + dx);
      }
      logits[((size_t)bh * SPAD + qi) * SPAD + kj] = acc[j][v] * 0.125f + m;
    }
}

// ---------------------------------------------------------------------------
// row softmax over first SEQ keys; emits bf16 probs, padded rows/cols -> 0
// ---------------------------------------------------------------------------
__global__ __launch_bounds__(256)
void softmax_k(const float* __restrict__ logits, unsigned short* __restrict__ attnbf)
{
  __shared__ float red[256];
  const int row = blockIdx.x;                // B*NH*SPAD
  const int s   = row % SPAD;
  const float* ar = logits + (size_t)row * SPAD;
  unsigned short* ab = attnbf + (size_t)row * SPAD;
  const int t = threadIdx.x;
  if (s >= SEQ) {                            // block-uniform
    for (int c = t; c < SPAD; c += 256) ab[c] = 0;
    return;
  }
  float v0 = (t < SEQ)       ? ar[t]       : -3.0e38f;
  float v1 = (t + 256 < SEQ) ? ar[t + 256] : -3.0e38f;
  red[t] = fmaxf(v0, v1); __syncthreads();
  for (int st = 128; st > 0; st >>= 1) { if (t < st) red[t] = fmaxf(red[t], red[t + st]); __syncthreads(); }
  const float mx = red[0]; __syncthreads();
  const float e0 = (t < SEQ)       ? expf(v0 - mx) : 0.0f;
  const float e1 = (t + 256 < SEQ) ? expf(v1 - mx) : 0.0f;
  red[t] = e0 + e1; __syncthreads();
  for (int st = 128; st > 0; st >>= 1) { if (t < st) red[t] += red[t + st]; __syncthreads(); }
  const float inv = 1.0f / red[0];
  ab[t] = (t < SEQ) ? f2bf(e0 * inv) : 0;
  const int c2 = t + 256;
  if (c2 < SPAD) ab[c2] = (c2 < SEQ) ? f2bf(e1 * inv) : 0;
}

// ---------------------------------------------------------------------------
// O[16 x 64] = probs[16 x SPAD] * V[SPAD x 64]; fragments from global (V^T).
// ---------------------------------------------------------------------------
__global__ __launch_bounds__(32)
void attn_av_k(const unsigned short* __restrict__ attn, const unsigned short* __restrict__ vT,
               unsigned short* __restrict__ obf)
{
  const int qt = blockIdx.x, bh = blockIdx.y;
  const int b = bh / NHH, hh = bh % NHH;
  const int lane = threadIdx.x;
  const int lrow = lane & 15, half = lane >> 4;
  const unsigned short* arow  = attn + ((size_t)bh * SPAD + qt * 16 + lrow) * SPAD;
  const unsigned short* vbase = vT + (size_t)bh * HDD * SPAD;

  v8f acc[4] = {};
  for (int kk = 0; kk < SPAD; kk += 32) {
    const v16bf af = load_afrag(arow + kk, half);
#pragma unroll
    for (int j = 0; j < 4; ++j) {
      const unsigned short* vp = vbase + (size_t)(j * 16 + lrow) * SPAD + kk + half * 16;
      acc[j] = wmma_bf16(af, load_bfrag(vp), acc[j]);
    }
  }
#pragma unroll
  for (int j = 0; j < 4; ++j)
#pragma unroll
    for (int v = 0; v < 8; ++v) {
      const int qi = qt * 16 + v + 8 * half;
      obf[((size_t)b * SPAD + qi) * DD + hh * HDD + j * 16 + lrow] = f2bf(acc[j][v]);
    }
}

// ---------------------------------------------------------------------------
// elementwise / reduction kernels
// ---------------------------------------------------------------------------
__global__ __launch_bounds__(256)
void layernorm_k(const float* __restrict__ x, const float* __restrict__ w,
                 const float* __restrict__ b, float* __restrict__ outf,
                 unsigned short* __restrict__ outbf, int fmt)
{
  __shared__ float red[256];
  const int row = blockIdx.x;
  const float* xr = x + (size_t)row * DD;
  const int t = threadIdx.x;
  float vals[4]; float s = 0.0f;
#pragma unroll
  for (int i = 0; i < 4; ++i) { vals[i] = xr[t + i * 256]; s += vals[i]; }
  red[t] = s; __syncthreads();
  for (int st = 128; st > 0; st >>= 1) { if (t < st) red[t] += red[t + st]; __syncthreads(); }
  const float mu = red[0] * (1.0f / DD); __syncthreads();
  s = 0.0f;
#pragma unroll
  for (int i = 0; i < 4; ++i) { const float d = vals[i] - mu; s += d * d; }
  red[t] = s; __syncthreads();
  for (int st = 128; st > 0; st >>= 1) { if (t < st) red[t] += red[t + st]; __syncthreads(); }
  const float rstd = rsqrtf(red[0] * (1.0f / DD) + 1e-5f);
#pragma unroll
  for (int i = 0; i < 4; ++i) {
    const int c = t + i * 256;
    const float o = (vals[i] - mu) * rstd * w[c] + b[c];
    if (fmt == 0) outf[(size_t)row * DD + c] = o;
    else          outbf[(size_t)row * DD + c] = f2bf(o);
  }
}

__global__ void patchify_k(const float* __restrict__ imgs, unsigned short* __restrict__ Ap)
{
  const int i = blockIdx.x * 256 + threadIdx.x;
  if (i >= BB * NTOK * KPATCH) return;
  const int col = i % KPATCH, tok = i / KPATCH;
  const int b = tok / NTOK, n = tok % NTOK;
  const int h = n / HPP, w = n % HPP;
  const int c = col / (PSZ * PSZ), pq = col % (PSZ * PSZ);
  const int p = pq / PSZ, qq = pq % PSZ;
  Ap[i] = f2bf(imgs[(((size_t)b * CCH + c) * IMGSZ + h * PSZ + p) * IMGSZ + w * PSZ + qq]);
}

__global__ void initx_k(const float* __restrict__ cls, const float* __restrict__ resid,
                        float* __restrict__ x)
{
  const int i = blockIdx.x * 256 + threadIdx.x;
  if (i >= BB * SPAD * DD) return;
  const int d = i % DD, rs = i / DD;
  const int b = rs / SPAD, s = rs % SPAD;
  float val = 0.0f;
  if (s == 0)         val = cls[d];
  else if (s <= NTOK) val = resid[((size_t)b * NTOK + (s - 1)) * DD + d];
  x[i] = val;
}

__global__ __launch_bounds__(256)
void pgsa_k(const float* __restrict__ xf, const float* __restrict__ resid,
            unsigned short* __restrict__ tokensbf)
{
  __shared__ float red[256];
  const int bd = blockIdx.x;                 // B*D channels
  const int b = bd / DD, d = bd % DD;
  const int n = threadIdx.x;                 // token 0..255
  const float t = xf[((size_t)b * SPAD + 1 + n) * DD + d];
  red[n] = t; __syncthreads();
  for (int st = 128; st > 0; st >>= 1) { if (n < st) red[n] += red[n + st]; __syncthreads(); }
  const float mu = red[0] * (1.0f / 256.0f); __syncthreads();
  const float xm2 = (t - mu) * (t - mu);
  red[n] = xm2; __syncthreads();
  for (int st = 128; st > 0; st >>= 1) { if (n < st) red[n] += red[n + st]; __syncthreads(); }
  const float y = xm2 / (4.0f * (red[0] * (1.0f / 255.0f) + 1e-4f)) + 0.5f;
  const float g = t * (1.0f / (1.0f + expf(-y)));
  tokensbf[((size_t)b * NTOK + n) * DD + d] = f2bf(g + resid[((size_t)b * NTOK + n) * DD + d]);
}

__global__ void unpatch_k(const float* __restrict__ pred, float* __restrict__ out)
{
  const int i = blockIdx.x * 256 + threadIdx.x;
  if (i >= BB * 2 * IMGSZ * IMGSZ) return;
  const int col = i % IMGSZ;
  int rest = i / IMGSZ;
  const int rowp = rest % IMGSZ; rest /= IMGSZ;
  const int ch = rest % 2, b = rest / 2;
  const int h = rowp / PSZ, p = rowp % PSZ;
  const int w = col / PSZ, q = col % PSZ;
  out[i] = pred[((size_t)b * NTOK + h * HPP + w) * PRED + (p * PSZ + q) * 2 + ch];
}

// ---------------------------------------------------------------------------
extern "C" void kernel_launch(void* const* d_in, const int* in_sizes, int n_in,
                              void* d_out, int out_size, void* d_ws, size_t ws_size,
                              hipStream_t stream)
{
  (void)in_sizes; (void)n_in; (void)out_size; (void)ws_size;
  const float* imgs    = (const float*)d_in[0];
  const float* patch_w = (const float*)d_in[1];
  const float* patch_b = (const float*)d_in[2];
  const float* cls     = (const float*)d_in[3];
  const float* ln1_w   = (const float*)d_in[4];
  const float* ln1_b   = (const float*)d_in[5];
  const float* wq      = (const float*)d_in[6];
  const float* wk      = (const float*)d_in[7];
  const float* wv      = (const float*)d_in[8];
  const float* bq      = (const float*)d_in[9];
  const float* bk      = (const float*)d_in[10];
  const float* bv      = (const float*)d_in[11];
  const float* wo      = (const float*)d_in[12];
  const float* bo      = (const float*)d_in[13];
  const float* ln2_w   = (const float*)d_in[14];
  const float* ln2_b   = (const float*)d_in[15];
  const float* w1      = (const float*)d_in[16];
  const float* b1      = (const float*)d_in[17];
  const float* w2      = (const float*)d_in[18];
  const float* b2      = (const float*)d_in[19];
  const float* normf_w = (const float*)d_in[20];
  const float* normf_b = (const float*)d_in[21];
  const float* enp_w   = (const float*)d_in[22];
  const float* enp_b   = (const float*)d_in[23];
  float* out = (float*)d_out;

  char* wsb = (char*)d_ws;
  size_t off = 0;
  auto alloc = [&](size_t bytes) {
    void* p = wsb + off;
    off += (bytes + 255) & ~(size_t)255;       // keep 256B alignment
    return p;
  };
  unsigned short* Apbf   = (unsigned short*)alloc((size_t)BB * NTOK * KPATCH * 2);
  float*          resid  = (float*)alloc((size_t)BB * NTOK * DD * 4);
  float*          x      = (float*)alloc((size_t)BB * SPAD * DD * 4);
  unsigned short* hbf    = (unsigned short*)alloc((size_t)BB * SPAD * DD * 2);
  unsigned short* qbf    = (unsigned short*)alloc((size_t)BB * SPAD * DD * 2);
  unsigned short* kbf    = (unsigned short*)alloc((size_t)BB * SPAD * DD * 2);
  unsigned short* vTbf   = (unsigned short*)alloc((size_t)BB * NHH * HDD * SPAD * 2);
  float*          logits = (float*)alloc((size_t)BB * NHH * SPAD * SPAD * 4);
  unsigned short* attnbf = (unsigned short*)alloc((size_t)BB * NHH * SPAD * SPAD * 2);
  unsigned short* obf    = (unsigned short*)alloc((size_t)BB * SPAD * DD * 2);
  unsigned short* midbf  = (unsigned short*)alloc((size_t)BB * SPAD * DFFF * 2);
  float*          xf     = (float*)alloc((size_t)BB * SPAD * DD * 4);
  unsigned short* tokbf  = (unsigned short*)alloc((size_t)BB * NTOK * DD * 2);
  float*          predf  = (float*)alloc((size_t)BB * NTOK * PRED * 4);

  const int Mx = BB * SPAD;                    // 2304 = 18 * 128

  patchify_k<<<(BB * NTOK * KPATCH + 255) / 256, 256, 0, stream>>>(imgs, Apbf);
  gemm_bf16_wmma<<<dim3(DD / 128, (BB * NTOK) / 128), 256, 0, stream>>>(
      Apbf, patch_w, patch_b, nullptr, resid, nullptr, BB * NTOK, DD, KPATCH, 0, 0, 0);
  initx_k<<<(BB * SPAD * DD + 255) / 256, 256, 0, stream>>>(cls, resid, x);

  const dim3 gq(DD / 128, Mx / 128);
  for (int l = 0; l < LLAY; ++l) {
    layernorm_k<<<Mx, 256, 0, stream>>>(x, ln1_w + l * DD, ln1_b + l * DD, nullptr, hbf, 1);
    gemm_bf16_wmma<<<gq, 256, 0, stream>>>(hbf, wq + (size_t)l * DD * DD, bq + l * DD,
                                           nullptr, nullptr, qbf, Mx, DD, DD, 0, 0, 1);
    gemm_bf16_wmma<<<gq, 256, 0, stream>>>(hbf, wk + (size_t)l * DD * DD, bk + l * DD,
                                           nullptr, nullptr, kbf, Mx, DD, DD, 0, 0, 1);
    gemm_bf16_wmma<<<gq, 256, 0, stream>>>(hbf, wv + (size_t)l * DD * DD, bv + l * DD,
                                           nullptr, nullptr, vTbf, Mx, DD, DD, 0, 0, 2);
    attn_logits_k<<<dim3(SPAD / 16, SPAD / 32, BB * NHH), 32, 0, stream>>>(qbf, kbf, logits);
    softmax_k<<<BB * NHH * SPAD, 256, 0, stream>>>(logits, attnbf);
    attn_av_k<<<dim3(SPAD / 16, BB * NHH), 32, 0, stream>>>(attnbf, vTbf, obf);
    gemm_bf16_wmma<<<gq, 256, 0, stream>>>(obf, wo + (size_t)l * DD * DD, bo + l * DD,
                                           x, x, nullptr, Mx, DD, DD, 0, 1, 0);
    layernorm_k<<<Mx, 256, 0, stream>>>(x, ln2_w + l * DD, ln2_b + l * DD, nullptr, hbf, 1);
    gemm_bf16_wmma<<<dim3(DFFF / 128, Mx / 128), 256, 0, stream>>>(
        hbf, w1 + (size_t)l * DD * DFFF, b1 + l * DFFF, nullptr, nullptr, midbf,
        Mx, DFFF, DD, 1, 0, 1);
    gemm_bf16_wmma<<<gq, 256, 0, stream>>>(midbf, w2 + (size_t)l * DFFF * DD, b2 + l * DD,
                                           x, x, nullptr, Mx, DD, DFFF, 0, 1, 0);
  }

  layernorm_k<<<Mx, 256, 0, stream>>>(x, normf_w, normf_b, xf, nullptr, 0);
  pgsa_k<<<BB * DD, 256, 0, stream>>>(xf, resid, tokbf);
  gemm_bf16_wmma<<<dim3(PRED / 128, (BB * NTOK) / 128), 256, 0, stream>>>(
      tokbf, enp_w, enp_b, nullptr, predf, nullptr, BB * NTOK, PRED, DD, 0, 0, 0);
  unpatch_k<<<(BB * 2 * IMGSZ * IMGSZ + 255) / 256, 256, 0, stream>>>(predf, out);
}
